// Capsule_Net_6373731467784
// MI455X (gfx1250) — matrix operand order repaired and checked by reading, not compiled, not run on hardware
//
#include <hip/hip_runtime.h>
#include <hip/hip_bf16.h>
#include <math.h>

// ---------------------------------------------------------------------------
// CapsNet forward for MI455X (gfx1250, wave32, WMMA).
// Heavy convs + decoder run as bf16 WMMA GEMMs (v_wmma_f32_16x16x32_bf16,
// f32 accumulate). pconv is an implicit GEMM reading bf16 NHWC activations.
// B panels are staged through LDS with GLOBAL_LOAD_ASYNC_TO_LDS_B128
// (ASYNCcnt-tracked, double buffered).
// ---------------------------------------------------------------------------

typedef __attribute__((ext_vector_type(16))) __bf16 v16bf;
typedef __attribute__((ext_vector_type(8)))  float  v8f;

union FragU {
    v16bf v;
    uint4 q[2];
};

#define B_SZ 256          // batch
#define NCAPS 1152
#define NOUT 10
#define DIN 8
#define DOUT 16

// Async 16-byte global->LDS copy (CDNA5 GLOBAL_LOAD_ASYNC_TO_LDS_B128).
// AS3->generic addrspacecast on AMDGPU is {aperture_hi, lds_offset}, so the
// low 32 bits of a generic pointer into __shared__ are exactly the
// workgroup-relative LDS byte address the instruction's VDST expects.
__device__ __forceinline__ void async_copy16(void* dst_shared, const void* src_global) {
    unsigned lds_off = (unsigned)(uintptr_t)dst_shared;
    asm volatile("global_load_async_to_lds_b128 %0, %1, off"
                 :: "v"(lds_off), "v"(src_global) : "memory");
}
__device__ __forceinline__ void wait_async0() {
    asm volatile("s_wait_asynccnt 0x0" ::: "memory");
}

// ---------------------------------------------------------------------------
// Weight repack: dst is (Npad, Kp) bf16, K-contiguous per row, zero padded.
// mode 0: src[n*K + k]          (conv1_w flattened (256,81))
// mode 1: src[k*N + n]          (decoder weights (K,N))
// mode 2: pconv_w (oc,ic,9,9):  k = (ky*9+kx)*256 + ic
// ---------------------------------------------------------------------------
__global__ void k_cast_bt(const float* __restrict__ src, __bf16* __restrict__ dst,
                          int N, int K, int Npad, int Kp, int mode) {
    long idx = (long)blockIdx.x * blockDim.x + threadIdx.x;
    long total = (long)Npad * Kp;
    if (idx >= total) return;
    int n = (int)(idx / Kp);
    int k = (int)(idx - (long)n * Kp);
    float v = 0.0f;
    if (n < N && k < K) {
        if (mode == 0)      v = src[(long)n * K + k];
        else if (mode == 1) v = src[(long)k * N + n];
        else {              // pconv: k -> (ky,kx,ic)
            int kk = k >> 8;       // ky*9+kx
            int ic = k & 255;
            v = src[((long)n * 256 + ic) * 81 + kk];
        }
    }
    dst[idx] = (__bf16)v;
}

// ---------------------------------------------------------------------------
// im2col for conv1: A1 (M=256*20*20, Kp=96) bf16, K=81 valid, rest zero.
// ---------------------------------------------------------------------------
__global__ void k_im2col1(const float* __restrict__ x, __bf16* __restrict__ A1) {
    long idx = (long)blockIdx.x * blockDim.x + threadIdx.x;
    const long total = 102400L * 96;
    if (idx >= total) return;
    int m = (int)(idx / 96);
    int k = (int)(idx - (long)m * 96);
    float v = 0.0f;
    if (k < 81) {
        int b = m / 400, r = m - b * 400;
        int oy = r / 20, ox = r - oy * 20;
        int ky = k / 9,  kx = k - ky * 9;
        v = x[(long)b * 784 + (oy + ky) * 28 + (ox + kx)];
    }
    A1[idx] = (__bf16)v;
}

// Implicit-GEMM address for pconv (stride-2 9x9 over bf16 NHWC (B,20,20,256)).
__device__ inline const __bf16* pconv_ptr(const __bf16* __restrict__ h1,
                                          int b, int oy, int ox, int k) {
    int kk = k >> 8;               // ky*9+kx
    int c  = k & 255;
    int ky = kk / 9, kx = kk - ky * 9;
    int iy = 2 * oy + ky, ix = 2 * ox + kx;
    return h1 + (((long)b * 20 + iy) * 20 + ix) * 256 + c;
}

// ---------------------------------------------------------------------------
// WMMA GEMM: C(MxN) = A(MxKp) * Bt(NxKp)^T, f32 accumulate.
// Block: 256 threads = 8 waves (4 M-waves x 2 N-waves), block tile 256x64,
// wave tile 64x32 (4x2 accumulators of 16x16). M must be a multiple of 256,
// Bt allocation padded to gridDim.y*64 rows, Kp multiple of 32.
// B panel (64 rows x 32 K = 4KB) is async-staged into LDS, double buffered:
// 256 lanes x one B128 async copy per K-step.
// AMODE 0: A row-major (M,Kp). AMODE 1: implicit pconv from NHWC h1.
// EPI 0: +bias -> bf16 | 1: +bias -> f32 | 2: +bias,ReLU -> bf16
// EPI 3: +bias,sigmoid -> f32
// ---------------------------------------------------------------------------
template<int AMODE, int EPI>
__global__ __launch_bounds__(256) void k_wmma_gemm(
        const __bf16* __restrict__ A, const __bf16* __restrict__ Bt,
        const float* __restrict__ bias, void* __restrict__ Cout,
        int M, int N, int Kp, int ldC) {
    __shared__ __bf16 sB[2][64 * 32];   // [stage][row*32 + k], 4KB per stage

    const int lane  = threadIdx.x & 31;
    const int wave  = threadIdx.x >> 5;
    const int waveM = wave & 3;
    const int waveN = wave >> 2;
    const int m_base = blockIdx.x * 256 + waveM * 64;
    const int n_tile = blockIdx.y * 64;
    const int n_base = n_tile + waveN * 32;
    const int half = lane >> 4;
    const int l15  = lane & 15;

    // Staging assignment: thread t copies 8 bf16 (16B) of B row (t>>2),
    // chunk (t&3), each K-step.
    const int srow   = threadIdx.x >> 2;   // 0..63
    const int schunk = threadIdx.x & 3;    // 0..3
    const __bf16* gB = Bt + (long)(n_tile + srow) * Kp + schunk * 8;
    __bf16* lB0 = &sB[0][srow * 32 + schunk * 8];
    __bf16* lB1 = &sB[1][srow * 32 + schunk * 8];

    v8f acc[4][2] = {};

    // Prime stage 0.
    async_copy16(lB0, gB);

    int stage = 0;
    for (int k0 = 0; k0 < Kp; k0 += 32) {
        wait_async0();
        __syncthreads();           // stage ready for all waves; safe to refill other
        if (k0 + 32 < Kp)
            async_copy16(stage ? lB0 : lB1, gB + (k0 + 32));

        // --- B fragments from LDS: lane = col; K chunks {0-7,16-23}/{8-15,24-31}
        FragU bfr[2];
#pragma unroll
        for (int j = 0; j < 2; ++j) {
            const __bf16* p = &sB[stage][(waveN * 32 + j * 16 + l15) * 32 + 8 * half];
            bfr[j].q[0] = *(const uint4*)(p);
            bfr[j].q[1] = *(const uint4*)(p + 16);
        }
        // --- A fragments: lane holds row m; K chunks {0-7,16-23}/{8-15,24-31}
        FragU afr[4];
#pragma unroll
        for (int i = 0; i < 4; ++i) {
            int row = m_base + i * 16 + l15;
            if (AMODE == 0) {
                const __bf16* p = A + (long)row * Kp + k0 + 8 * half;
                afr[i].q[0] = *(const uint4*)(p);
                afr[i].q[1] = *(const uint4*)(p + 16);
            } else {
                int b = row / 36, r = row - b * 36;
                int oy = r / 6,  ox = r - oy * 6;
                int kc = k0 + 8 * half;
                afr[i].q[0] = *(const uint4*)pconv_ptr(A, b, oy, ox, kc);
                afr[i].q[1] = *(const uint4*)pconv_ptr(A, b, oy, ox, kc + 16);
            }
        }
#pragma unroll
        for (int i = 0; i < 4; ++i)
#pragma unroll
            for (int j = 0; j < 2; ++j)
                acc[i][j] = __builtin_amdgcn_wmma_f32_16x16x32_bf16(
                    false, afr[i].v, false, bfr[j].v,
                    (short)0, acc[i][j], false, false);
        stage ^= 1;
    }

    // Epilogue: D layout -> VGPR r holds row r+8*half, col = lane&15.
#pragma unroll
    for (int j = 0; j < 2; ++j) {
        int col = n_base + j * 16 + l15;
        if (col >= N) continue;
        float bv = bias ? bias[col] : 0.0f;
#pragma unroll
        for (int i = 0; i < 4; ++i) {
#pragma unroll
            for (int r = 0; r < 8; ++r) {
                int row = m_base + i * 16 + r + 8 * half;
                float v = acc[i][j][r] + bv;
                long off = (long)row * ldC + col;
                if (EPI == 0)      ((__bf16*)Cout)[off] = (__bf16)v;
                else if (EPI == 1) ((float*)Cout)[off] = v;
                else if (EPI == 2) ((__bf16*)Cout)[off] = (__bf16)fmaxf(v, 0.0f);
                else               ((float*)Cout)[off] = 1.0f / (1.0f + __expf(-v));
            }
        }
    }
}

// ---------------------------------------------------------------------------
// NHWC (b,oy,ox,oc) f32 -> NCHW-flat (b, oc*36+oy*6+ox) f32.
// ---------------------------------------------------------------------------
__global__ void k_gather_nchw(const float* __restrict__ h2, float* __restrict__ u_nchw) {
    long idx = (long)blockIdx.x * blockDim.x + threadIdx.x;
    const long total = (long)B_SZ * 9216;
    if (idx >= total) return;
    int n = (int)(idx / 9216);
    int f = (int)(idx - (long)n * 9216);
    int oc = f / 36, r = f - oc * 36;
    int oy = r / 6,  ox = r - oy * 6;
    u_nchw[idx] = h2[((long)n * 36 + oy * 6 + ox) * 256 + oc];
}

// squash over capsules of dim 8: x * |x| / (1+|x|^2)
__global__ void k_squash_u(const float* __restrict__ u_nchw, float* __restrict__ u_in) {
    int cap = blockIdx.x * blockDim.x + threadIdx.x;
    if (cap >= B_SZ * NCAPS) return;
    const float* p = u_nchw + (long)cap * DIN;
    float n2 = 0.0f;
#pragma unroll
    for (int d = 0; d < DIN; ++d) n2 += p[d] * p[d];
    float scale = sqrtf(n2) / (1.0f + n2);
#pragma unroll
    for (int d = 0; d < DIN; ++d) u_in[(long)cap * DIN + d] = p[d] * scale;
}

// u_hat[n,i,j,:] = W_caps[i,j] (16x8) @ u_in[n,i,:]  -> bf16
__global__ void k_uhat(const float* __restrict__ u_in, const float* __restrict__ W,
                       __bf16* __restrict__ u) {
    long t = (long)blockIdx.x * blockDim.x + threadIdx.x;
    const long total = (long)B_SZ * NCAPS * NOUT;
    if (t >= total) return;
    int n = (int)(t / (NCAPS * NOUT));
    int rem = (int)(t - (long)n * NCAPS * NOUT);
    int i = rem / NOUT, j = rem - i * NOUT;
    const float* uin = u_in + ((long)n * NCAPS + i) * DIN;
    const float* Wp  = W + (long)(i * NOUT + j) * (DOUT * DIN);
    __bf16* out = u + ((long)(n * NCAPS + i) * NOUT + j) * DOUT;
#pragma unroll
    for (int a = 0; a < DOUT; ++a) {
        float s = 0.0f;
#pragma unroll
        for (int d = 0; d < DIN; ++d) s += Wp[a * DIN + d] * uin[d];
        out[a] = (__bf16)s;
    }
}

__global__ void k_zero_f32(float* __restrict__ p, int n) {
    int i = blockIdx.x * blockDim.x + threadIdx.x;
    if (i < n) p[i] = 0.0f;
}

__global__ void k_softmax(const float* __restrict__ b, float* __restrict__ c) {
    int i = blockIdx.x * blockDim.x + threadIdx.x;
    if (i >= NCAPS) return;
    float m = -1e30f;
#pragma unroll
    for (int j = 0; j < NOUT; ++j) m = fmaxf(m, b[i * NOUT + j]);
    float e[NOUT], s = 0.0f;
#pragma unroll
    for (int j = 0; j < NOUT; ++j) { e[j] = __expf(b[i * NOUT + j] - m); s += e[j]; }
    float inv = 1.0f / s;
#pragma unroll
    for (int j = 0; j < NOUT; ++j) c[i * NOUT + j] = e[j] * inv;
}

// s[n,j,:] = sum_i c[i,j]*u[n,i,j,:]; v = squash(s). One block per (n,j).
__global__ __launch_bounds__(256) void k_s_v(const __bf16* __restrict__ u,
                                             const float* __restrict__ cmat,
                                             float* __restrict__ v_out) {
    __shared__ float red[256][DOUT];
    int n = blockIdx.x / NOUT;
    int j = blockIdx.x - n * NOUT;
    int tid = threadIdx.x;
    float ps[DOUT];
#pragma unroll
    for (int a = 0; a < DOUT; ++a) ps[a] = 0.0f;
    for (int i = tid; i < NCAPS; i += 256) {
        float cij = cmat[i * NOUT + j];
        const __bf16* up = u + ((long)(n * NCAPS + i) * NOUT + j) * DOUT;
#pragma unroll
        for (int a = 0; a < DOUT; ++a) ps[a] += cij * (float)up[a];
    }
#pragma unroll
    for (int a = 0; a < DOUT; ++a) red[tid][a] = ps[a];
    __syncthreads();
    for (int s2 = 128; s2 > 0; s2 >>= 1) {
        if (tid < s2)
#pragma unroll
            for (int a = 0; a < DOUT; ++a) red[tid][a] += red[tid + s2][a];
        __syncthreads();
    }
    if (tid == 0) {
        float n2 = 0.0f;
#pragma unroll
        for (int a = 0; a < DOUT; ++a) n2 += red[0][a] * red[0][a];
        float scale = sqrtf(n2) / (1.0f + n2);
#pragma unroll
        for (int a = 0; a < DOUT; ++a)
            v_out[((long)n * NOUT + j) * DOUT + a] = red[0][a] * scale;
    }
}

// b[i,j] += sum_n sum_a u[n,i,j,a]*v[n,j,a]
__global__ void k_b_update(const __bf16* __restrict__ u, const float* __restrict__ v,
                           float* __restrict__ b) {
    int t = blockIdx.x * blockDim.x + threadIdx.x;
    if (t >= NCAPS * NOUT) return;
    int i = t / NOUT, j = t - i * NOUT;
    float acc = 0.0f;
    for (int n = 0; n < B_SZ; ++n) {
        const __bf16* up = u + ((long)(n * NCAPS + i) * NOUT + j) * DOUT;
        const float*  vp = v + ((long)n * NOUT + j) * DOUT;
#pragma unroll
        for (int a = 0; a < DOUT; ++a) acc += (float)up[a] * vp[a];
    }
    b[t] += acc;
}

// argmax(|v_j|) mask -> recon0 (256,160) bf16
__global__ void k_mask(const float* __restrict__ v, __bf16* __restrict__ recon0) {
    int n = threadIdx.x;   // one block of 256
    const float* vp = v + (long)n * (NOUT * DOUT);
    int best = 0; float bestn = -1.0f;
#pragma unroll
    for (int j = 0; j < NOUT; ++j) {
        float s = 0.0f;
#pragma unroll
        for (int a = 0; a < DOUT; ++a) s += vp[j * DOUT + a] * vp[j * DOUT + a];
        if (s > bestn) { bestn = s; best = j; }
    }
#pragma unroll
    for (int j = 0; j < NOUT; ++j)
#pragma unroll
        for (int a = 0; a < DOUT; ++a)
            recon0[(long)n * 160 + j * DOUT + a] =
                (__bf16)((j == best) ? vp[j * DOUT + a] : 0.0f);
}

// ---------------------------------------------------------------------------
// Host driver
// ---------------------------------------------------------------------------
static inline int cdiv(long a, long b) { return (int)((a + b - 1) / b); }

extern "C" void kernel_launch(void* const* d_in, const int* in_sizes, int n_in,
                              void* d_out, int out_size, void* d_ws, size_t ws_size,
                              hipStream_t stream) {
    const float* x       = (const float*)d_in[0];
    const float* conv1_w = (const float*)d_in[1];
    const float* conv1_b = (const float*)d_in[2];
    const float* pconv_w = (const float*)d_in[3];
    const float* pconv_b = (const float*)d_in[4];
    const float* W_caps  = (const float*)d_in[5];
    const float* dec_w1  = (const float*)d_in[6];
    const float* dec_b1  = (const float*)d_in[7];
    const float* dec_w2  = (const float*)d_in[8];
    const float* dec_b2  = (const float*)d_in[9];
    const float* dec_w3  = (const float*)d_in[10];
    const float* dec_b3  = (const float*)d_in[11];
    float* out = (float*)d_out;   // [0,40960): v ; [40960, +200704): recon

    // Workspace carve-up (all sizes multiple of 256 bytes)
    char* p = (char*)d_ws;
    __bf16* A1    = (__bf16*)p; p += 102400L * 96 * 2;     // conv1 im2col
    __bf16* Bt1   = (__bf16*)p; p += 256L * 96 * 2;        // conv1 weights
    __bf16* h1    = (__bf16*)p; p += 102400L * 256 * 2;    // conv1 out, NHWC bf16
    __bf16* Bt2   = (__bf16*)p; p += 256L * 20736 * 2;     // pconv weights
    float*  h2    = (float*)p;  p += 9216L * 256 * 4;      // pconv out, NHWC f32
    float*  u_nc  = (float*)p;  p += 256L * 9216 * 4;      // NCHW-flat
    float*  u_in  = (float*)p;  p += 256L * 9216 * 4;      // squashed caps
    __bf16* u_hat = (__bf16*)p; p += 256L * 1152 * 160 * 2;
    float*  bmat  = (float*)p;  p += 11520L * 4;
    float*  cmat  = (float*)p;  p += 11520L * 4;
    __bf16* rec0  = (__bf16*)p; p += 256L * 160 * 2;
    __bf16* act1  = (__bf16*)p; p += 256L * 512 * 2;
    __bf16* act2  = (__bf16*)p; p += 256L * 1024 * 2;
    __bf16* Btd1  = (__bf16*)p; p += 512L * 160 * 2;
    __bf16* Btd2  = (__bf16*)p; p += 1024L * 512 * 2;
    __bf16* Btd3  = (__bf16*)p; p += 832L * 1024 * 2;      // 784 padded to 832

    // --- weight repacks + im2col -------------------------------------------
    k_cast_bt<<<cdiv(256L * 96, 256), 256, 0, stream>>>(conv1_w, Bt1, 256, 81, 256, 96, 0);
    k_cast_bt<<<cdiv(256L * 20736, 256), 256, 0, stream>>>(pconv_w, Bt2, 256, 20736, 256, 20736, 2);
    k_cast_bt<<<cdiv(512L * 160, 256), 256, 0, stream>>>(dec_w1, Btd1, 512, 160, 512, 160, 1);
    k_cast_bt<<<cdiv(1024L * 512, 256), 256, 0, stream>>>(dec_w2, Btd2, 1024, 512, 1024, 512, 1);
    k_cast_bt<<<cdiv(832L * 1024, 256), 256, 0, stream>>>(dec_w3, Btd3, 784, 1024, 832, 1024, 1);
    k_im2col1<<<cdiv(102400L * 96, 256), 256, 0, stream>>>(x, A1);

    // --- conv1 as GEMM: (102400 x 96) * (96 x 256) -> h1 bf16 NHWC ---------
    k_wmma_gemm<0, 0><<<dim3(400, 4), 256, 0, stream>>>(A1, Bt1, conv1_b, h1,
                                                        102400, 256, 96, 256);
    // --- pconv implicit GEMM: (9216 x 20736) * (20736 x 256) -> h2 f32 -----
    k_wmma_gemm<1, 1><<<dim3(36, 4), 256, 0, stream>>>(h1, Bt2, pconv_b, h2,
                                                       9216, 256, 20736, 256);

    // --- capsule prep -------------------------------------------------------
    k_gather_nchw<<<cdiv(256L * 9216, 256), 256, 0, stream>>>(h2, u_nc);
    k_squash_u<<<cdiv(256L * 1152, 256), 256, 0, stream>>>(u_nc, u_in);
    k_uhat<<<cdiv(256L * 1152 * 10, 256), 256, 0, stream>>>(u_in, W_caps, u_hat);

    // --- routing by agreement (b shared across batch) ----------------------
    k_zero_f32<<<cdiv(11520, 256), 256, 0, stream>>>(bmat, 11520);
    for (int r = 0; r < 4; ++r) {
        k_softmax<<<cdiv(1152, 256), 256, 0, stream>>>(bmat, cmat);
        k_s_v<<<256 * 10, 256, 0, stream>>>(u_hat, cmat, out);   // v -> d_out[0:40960)
        if (r < 3)
            k_b_update<<<cdiv(11520, 256), 256, 0, stream>>>(u_hat, out, bmat);
    }

    // --- decoder ------------------------------------------------------------
    k_mask<<<1, 256, 0, stream>>>(out, rec0);
    k_wmma_gemm<0, 2><<<dim3(1, 8), 256, 0, stream>>>(rec0, Btd1, dec_b1, act1,
                                                      256, 512, 160, 512);
    k_wmma_gemm<0, 2><<<dim3(1, 16), 256, 0, stream>>>(act1, Btd2, dec_b2, act2,
                                                       256, 1024, 512, 1024);
    k_wmma_gemm<0, 3><<<dim3(1, 13), 256, 0, stream>>>(act2, Btd3, dec_b3, out + 40960,
                                                       256, 784, 1024, 784);
    (void)in_sizes; (void)n_in; (void)out_size; (void)ws_size;
}
